// MultiHeadLocationAwareAttention_44367012168178
// MI455X (gfx1250) — matrix operand order repaired and checked by reading, not compiled, not run on hardware
//
#include <hip/hip_runtime.h>
#include <hip/hip_bf16.h>
#include <math.h>

// Problem constants (from reference)
#define B_    32
#define T_    2000
#define ENC_  1024
#define DEC_  1024
#define ATTN_ 512
#define OUT_  1024
#define H_    4
#define C_    64
#define KS_   50
#define KLEN_ 101
#define PH_   256
#define TS_   8          // T-split for ctx partials
#define NK_   (ENC_ / 32)

typedef __bf16 bf16_t;
typedef __attribute__((ext_vector_type(16))) __bf16 v16bf;
typedef __attribute__((ext_vector_type(8)))  __bf16 bf16x8;
typedef __attribute__((ext_vector_type(8)))  float  v8f;
typedef __attribute__((ext_vector_type(4)))  float  f32x4;

union Frag16 { v16bf v; bf16x8 h[2]; };

// gfx1250 async global->LDS copy (GVS mode: SGPR64 base + VGPR32 offset).
// vdst VGPR carries the per-lane LDS byte offset (generic ptr truncated).
__device__ __forceinline__ void async_load_b128(uint32_t lds_off,
                                                uint32_t voff,
                                                const void* sbase) {
  asm volatile("global_load_async_to_lds_b128 %0, %1, %2"
               :: "v"(lds_off), "v"(voff), "s"(sbase)
               : "memory");
}
__device__ __forceinline__ void wait_asynccnt_le4() {
  asm volatile("s_wait_asynccnt 4" ::: "memory");
}
__device__ __forceinline__ void wait_asynccnt_0() {
  asm volatile("s_wait_asynccnt 0" ::: "memory");
}

// ---------------------------------------------------------------------------
// K0: Wenc fp32 -> bf16 copy (makes B operand byte-copyable for async DMA)
// ---------------------------------------------------------------------------
__global__ __launch_bounds__(256) void wencbf_kernel(
    const float* __restrict__ Wenc, bf16_t* __restrict__ WencBF) {
  size_t i = ((size_t)blockIdx.x * 256 + threadIdx.x) * 8;
  f32x4 v0 = *(const f32x4*)(Wenc + i);
  f32x4 v1 = *(const f32x4*)(Wenc + i + 4);
  bf16x8 pk;
  pk[0] = (bf16_t)v0.x; pk[1] = (bf16_t)v0.y;
  pk[2] = (bf16_t)v0.z; pk[3] = (bf16_t)v0.w;
  pk[4] = (bf16_t)v1.x; pk[5] = (bf16_t)v1.y;
  pk[6] = (bf16_t)v1.z; pk[7] = (bf16_t)v1.w;
  *(bf16x8*)(WencBF + i) = pk;
}

// ---------------------------------------------------------------------------
// K1: prefix sums of Wconv taps: cumW[hc][p] = sum_{k<p} Wconv[hc,0,k]
// (prev_attn is exactly uniform, so the location conv collapses to this)
// ---------------------------------------------------------------------------
__global__ void cumw_kernel(const float* __restrict__ Wconv,
                            float* __restrict__ cumW) {
  int hc = blockIdx.x * blockDim.x + threadIdx.x;
  if (hc >= H_ * C_) return;
  const float* w = Wconv + (size_t)hc * KLEN_;
  float* o = cumW + (size_t)hc * (KLEN_ + 1);
  float s = 0.0f;
  o[0] = 0.0f;
  for (int k = 0; k < KLEN_; ++k) { s += w[k]; o[k + 1] = s; }
}

// ---------------------------------------------------------------------------
// K2: Mtab[h,a,p] = sum_c cumW[h*C+c][p] * Wloc[h,a,c]
// ---------------------------------------------------------------------------
__global__ __launch_bounds__(128) void mtab_kernel(
    const float* __restrict__ cumW, const float* __restrict__ Wloc,
    float* __restrict__ Mtab) {
  int ha = blockIdx.x;          // h*ATTN + a
  int h  = ha >> 9;             // ATTN = 512
  int p  = threadIdx.x;
  if (p >= KLEN_ + 1) return;
  const float* wl = Wloc + (size_t)ha * C_;
  float s = 0.0f;
  for (int c = 0; c < C_; ++c)
    s += cumW[(size_t)(h * C_ + c) * (KLEN_ + 1) + p] * wl[c];
  Mtab[(size_t)ha * (KLEN_ + 1) + p] = s;
}

// ---------------------------------------------------------------------------
// K3: dec_h[h,b,a] = Wdec[h,a,:] . dec_states[b,:] + bdec[h,a]
// ---------------------------------------------------------------------------
__global__ __launch_bounds__(256) void dech_kernel(
    const float* __restrict__ dec_states, const float* __restrict__ Wdec,
    const float* __restrict__ bdec, float* __restrict__ decH) {
  __shared__ float ds[DEC_];
  int hb = blockIdx.x;
  int h = hb / B_, b = hb % B_;
  for (int d = threadIdx.x; d < DEC_; d += 256)
    ds[d] = dec_states[(size_t)b * DEC_ + d];
  __syncthreads();
  for (int a = threadIdx.x; a < ATTN_; a += 256) {
    const float* w = Wdec + ((size_t)(h * ATTN_) + a) * DEC_;
    float s = bdec[h * ATTN_ + a];
    for (int d = 0; d < DEC_; ++d) s += w[d] * ds[d];
    decH[((size_t)(h * B_) + b) * ATTN_ + a] = s;
  }
}

// ---------------------------------------------------------------------------
// K4: main WMMA kernel, double-buffered with async global->LDS for B.
// grid = (ceil(T/64), B, H*2); block = 256 (8 waves).
// WG tile: 64 t-rows x 256 attn-cols for one (b, h, n-half). K chunks of 32.
// Wave layout: waveM = wave&1 (32 rows), waveN = wave>>1 (64 cols).
// Writes partial score sums (over its 256 a's) to scoreP[nh,h,b,t].
// ---------------------------------------------------------------------------
__global__ __launch_bounds__(256) void scores_kernel(
    const float*  __restrict__ enc,      // [B,T,ENC] f32
    const int*    __restrict__ enc_len,  // [B]
    const bf16_t* __restrict__ WencBF,   // [H,ATTN,ENC] bf16
    const float*  __restrict__ benc,     // [H,ATTN]
    const float*  __restrict__ bloc,     // [H,ATTN]
    const float*  __restrict__ wattn,    // [H,ATTN]
    const float*  __restrict__ decH,     // [H,B,ATTN]
    const float*  __restrict__ Mtab,     // [H,ATTN,KLEN+1]
    float* __restrict__ scoreP)          // [2,H,B,T]
{
  __shared__ __align__(16) bf16_t As[2][64][40];   // 2 x (64 x 32, padded)
  __shared__ __align__(16) bf16_t Bs[2][256][40];  // 2 x (256 x 32, padded)
  __shared__ float biasS[256];
  __shared__ float wattnS[256];
  __shared__ float scoreS[64];

  const int tid   = threadIdx.x;
  const int lane  = tid & 31;
  const int wv    = tid >> 5;
  const int waveM = wv & 1;
  const int waveN = wv >> 1;              // 0..3
  const int t0 = blockIdx.x * 64;
  const int b  = blockIdx.y;
  const int h  = blockIdx.z >> 1;
  const int nh = blockIdx.z & 1;
  const int n0 = nh * 256;
  const int L  = enc_len[b];
  const float invL = 1.0f / (float)L;

  {
    int a = n0 + tid;                     // exactly 256 threads
    biasS[tid]  = decH[((size_t)(h * B_) + b) * ATTN_ + a] +
                  benc[h * ATTN_ + a] + bloc[h * ATTN_ + a];
    wattnS[tid] = wattn[h * ATTN_ + a];
  }
  if (tid < 64) scoreS[tid] = 0.0f;

  v8f acc[2][4];
  const v8f vzero = {0.f, 0.f, 0.f, 0.f, 0.f, 0.f, 0.f, 0.f};
#pragma unroll
  for (int mi = 0; mi < 2; ++mi)
#pragma unroll
    for (int ni = 0; ni < 4; ++ni) acc[mi][ni] = vzero;

  const float*  encB = enc + (size_t)b * T_ * ENC_;
  const bf16_t* WB   = WencBF + ((size_t)h * ATTN_ + n0) * ENC_;

  // per-thread staging geometry
  const int ar  = tid >> 2;               // A row 0..63
  const int acg = tid & 3;                // A col-group of 8
  const int brg = lane >> 2;              // B row-in-group 0..7
  const int bch = lane & 3;               // B 16-byte chunk 0..3

  // ---- prologue: stage buffer 0 ----
  {
    int t = t0 + ar;
    f32x4 v0 = {0.f,0.f,0.f,0.f}, v1 = {0.f,0.f,0.f,0.f};
    if (t < T_) {
      const float* p = encB + (size_t)t * ENC_ + acg * 8;
      v0 = *(const f32x4*)p; v1 = *(const f32x4*)(p + 4);
    }
    bf16x8 pk;
    pk[0]=(bf16_t)v0.x; pk[1]=(bf16_t)v0.y; pk[2]=(bf16_t)v0.z; pk[3]=(bf16_t)v0.w;
    pk[4]=(bf16_t)v1.x; pk[5]=(bf16_t)v1.y; pk[6]=(bf16_t)v1.z; pk[7]=(bf16_t)v1.w;
    *(bf16x8*)&As[0][ar][acg * 8] = pk;
#pragma unroll
    for (int i = 0; i < 4; ++i) {
      int aRow = (wv * 4 + i) * 8 + brg;  // 0..255
      uint32_t lds  = (uint32_t)(uintptr_t)&Bs[0][aRow][bch * 8];
      uint32_t voff = (uint32_t)(aRow * (ENC_ * 2) + bch * 16);
      async_load_b128(lds, voff, (const void*)WB);
    }
  }

  for (int it = 0; it < NK_; ++it) {
    const int cur = it & 1, nxt = cur ^ 1;
    if (it + 1 < NK_) {
      // ---- stage next buffer while current tile is consumed ----
      int k0n = (it + 1) * 32;
      int t = t0 + ar;
      f32x4 v0 = {0.f,0.f,0.f,0.f}, v1 = {0.f,0.f,0.f,0.f};
      if (t < T_) {
        const float* p = encB + (size_t)t * ENC_ + k0n + acg * 8;
        v0 = *(const f32x4*)p; v1 = *(const f32x4*)(p + 4);
      }
      bf16x8 pk;
      pk[0]=(bf16_t)v0.x; pk[1]=(bf16_t)v0.y; pk[2]=(bf16_t)v0.z; pk[3]=(bf16_t)v0.w;
      pk[4]=(bf16_t)v1.x; pk[5]=(bf16_t)v1.y; pk[6]=(bf16_t)v1.z; pk[7]=(bf16_t)v1.w;
      *(bf16x8*)&As[nxt][ar][acg * 8] = pk;
#pragma unroll
      for (int i = 0; i < 4; ++i) {
        int aRow = (wv * 4 + i) * 8 + brg;
        uint32_t lds  = (uint32_t)(uintptr_t)&Bs[nxt][aRow][bch * 8];
        uint32_t voff = (uint32_t)(aRow * (ENC_ * 2) + k0n * 2 + bch * 16);
        async_load_b128(lds, voff, (const void*)WB);
      }
      wait_asynccnt_le4();   // in-order: <=4 outstanding => current tile done
    } else {
      wait_asynccnt_0();
    }
    __syncthreads();         // all waves' cur-buffer data visible

    // ---- fragments (ISA 16-bit A 16x32 / B 32x16 lane layouts) ----
    Frag16 afr[2];
    {
      int row0  = waveM * 32 + (lane & 15);
      int base0 = (lane & 16) >> 1;        // 0 or 8
      afr[0].h[0] = *(const bf16x8*)&As[cur][row0][base0];
      afr[0].h[1] = *(const bf16x8*)&As[cur][row0][base0 + 16];
      afr[1].h[0] = *(const bf16x8*)&As[cur][row0 + 16][base0];
      afr[1].h[1] = *(const bf16x8*)&As[cur][row0 + 16][base0 + 16];
    }
    Frag16 bfr[4];
#pragma unroll
    for (int ni = 0; ni < 4; ++ni) {
      int brow = waveN * 64 + ni * 16 + (lane & 15);
      int kb   = (lane & 16) >> 1;         // 0 or 8
      bfr[ni].h[0] = *(const bf16x8*)&Bs[cur][brow][kb];
      bfr[ni].h[1] = *(const bf16x8*)&Bs[cur][brow][kb + 16];
    }
#pragma unroll
    for (int ni = 0; ni < 4; ++ni) {
      acc[0][ni] = __builtin_amdgcn_wmma_f32_16x16x32_bf16(
          false, afr[0].v, false, bfr[ni].v, (short)0, acc[0][ni], false, false);
      acc[1][ni] = __builtin_amdgcn_wmma_f32_16x16x32_bf16(
          false, afr[1].v, false, bfr[ni].v, (short)0, acc[1][ni], false, false);
    }
    __syncthreads();         // reads of cur done -> safe to overwrite next iter
  }

  // ---- fused epilogue: bias + loc(Mtab) + tanh, dot with wattn, reduce ----
#pragma unroll
  for (int mi = 0; mi < 2; ++mi) {
#pragma unroll
    for (int v = 0; v < 8; ++v) {
      int tl = waveM * 32 + mi * 16 + v + ((lane & 16) ? 8 : 0);
      int t  = t0 + tl;
      int p0 = KS_ - t;     p0 = p0 < 0 ? 0 : (p0 > KLEN_ ? KLEN_ : p0);
      int p1 = KS_ - t + L; p1 = p1 < 0 ? 0 : (p1 > KLEN_ ? KLEN_ : p1);
      float partial = 0.0f;
#pragma unroll
      for (int ni = 0; ni < 4; ++ni) {
        int al = waveN * 64 + ni * 16 + (lane & 15);     // local a (0..255)
        int a  = n0 + al;
        const float* mrow = Mtab + ((size_t)(h * ATTN_) + a) * (KLEN_ + 1);
        float loc = (mrow[p1] - mrow[p0]) * invL;
        float x = acc[mi][ni][v] + biasS[al] + loc;
        partial += tanhf(x) * wattnS[al];
      }
      partial += __shfl_xor(partial, 1, 32);
      partial += __shfl_xor(partial, 2, 32);
      partial += __shfl_xor(partial, 4, 32);
      partial += __shfl_xor(partial, 8, 32);
      if ((lane & 15) == 0) atomicAdd(&scoreS[tl], partial);
    }
  }
  __syncthreads();
  if (tid < 64) {
    int t = t0 + tid;
    if (t < T_)
      scoreP[((size_t)((nh * H_ + h) * B_) + b) * T_ + t] = scoreS[tid];
  }
}

// ---------------------------------------------------------------------------
// K5: masked softmax over T per (h,b); sums the two N-half partials
// ---------------------------------------------------------------------------
__global__ __launch_bounds__(256) void softmax_kernel(
    const float* __restrict__ scoreP, const int* __restrict__ enc_len,
    float* __restrict__ attn) {
  __shared__ float red[256];
  __shared__ float srow[T_];
  int hb = blockIdx.x;
  int b = hb % B_;
  int L = enc_len[b];
  const float* s0 = scoreP + (size_t)hb * T_;
  const float* s1 = scoreP + (size_t)(H_ * B_ + hb) * T_;
  for (int t = threadIdx.x; t < T_; t += 256) srow[t] = s0[t] + s1[t];
  __syncthreads();
  float m = -1e30f;
  for (int t = threadIdx.x; t < L; t += 256) m = fmaxf(m, srow[t]);
  red[threadIdx.x] = m; __syncthreads();
  for (int o = 128; o; o >>= 1) {
    if (threadIdx.x < o)
      red[threadIdx.x] = fmaxf(red[threadIdx.x], red[threadIdx.x + o]);
    __syncthreads();
  }
  m = red[0]; __syncthreads();
  float sum = 0.0f;
  for (int t = threadIdx.x; t < L; t += 256) sum += __expf(srow[t] - m);
  red[threadIdx.x] = sum; __syncthreads();
  for (int o = 128; o; o >>= 1) {
    if (threadIdx.x < o) red[threadIdx.x] += red[threadIdx.x + o];
    __syncthreads();
  }
  float inv = 1.0f / red[0];
  float* o = attn + (size_t)hb * T_;
  for (int t = threadIdx.x; t < T_; t += 256)
    o[t] = (t < L) ? __expf(srow[t] - m) * inv : 0.0f;
}

// ---------------------------------------------------------------------------
// K6: attn_weight[b,t] = mean over heads
// ---------------------------------------------------------------------------
__global__ void attnmean_kernel(const float* __restrict__ attn,
                                float* __restrict__ out) {
  int i = blockIdx.x * 256 + threadIdx.x;
  if (i >= B_ * T_) return;
  int b = i / T_, t = i % T_;
  float s = 0.0f;
  for (int hh = 0; hh < H_; ++hh)
    s += attn[((size_t)(hh * B_) + b) * T_ + t];
  out[i] = s * (1.0f / H_);
}

// ---------------------------------------------------------------------------
// K7: ctx partials over T-chunks: ctxP[ts,h,b,e] = sum_t attn*enc
// ---------------------------------------------------------------------------
__global__ __launch_bounds__(256) void ctx_kernel(
    const float* __restrict__ enc, const float* __restrict__ attn,
    float* __restrict__ ctxP) {
  __shared__ float at[H_][T_ / TS_];
  int e  = blockIdx.x * 256 + threadIdx.x;
  int b  = blockIdx.y;
  int ts = blockIdx.z;
  int tbeg = ts * (T_ / TS_);
  for (int i = threadIdx.x; i < H_ * (T_ / TS_); i += 256) {
    int hh = i / (T_ / TS_), tt = i % (T_ / TS_);
    at[hh][tt] = attn[((size_t)(hh * B_) + b) * T_ + tbeg + tt];
  }
  __syncthreads();
  float c0 = 0.f, c1 = 0.f, c2 = 0.f, c3 = 0.f;
  const float* ep = enc + ((size_t)b * T_ + tbeg) * ENC_ + e;
  for (int tt = 0; tt < T_ / TS_; ++tt) {
    float ev = ep[(size_t)tt * ENC_];
    c0 += at[0][tt] * ev; c1 += at[1][tt] * ev;
    c2 += at[2][tt] * ev; c3 += at[3][tt] * ev;
  }
  ctxP[(((size_t)ts * H_ + 0) * B_ + b) * ENC_ + e] = c0;
  ctxP[(((size_t)ts * H_ + 1) * B_ + b) * ENC_ + e] = c1;
  ctxP[(((size_t)ts * H_ + 2) * B_ + b) * ENC_ + e] = c2;
  ctxP[(((size_t)ts * H_ + 3) * B_ + b) * ENC_ + e] = c3;
}

// ---------------------------------------------------------------------------
// K8: out[b, h*PH+o] = ctx[h,b,:] . Wout[h,o,:] + bout[h,o]
// ---------------------------------------------------------------------------
__global__ __launch_bounds__(256) void out_kernel(
    const float* __restrict__ ctxP, const float* __restrict__ Wout,
    const float* __restrict__ bout, float* __restrict__ context) {
  __shared__ float cs[ENC_];
  int h = blockIdx.x, b = blockIdx.y;
#pragma unroll
  for (int j = 0; j < 4; ++j) {
    int e = threadIdx.x + j * 256;
    float s = 0.0f;
    for (int ts = 0; ts < TS_; ++ts)
      s += ctxP[(((size_t)ts * H_ + h) * B_ + b) * ENC_ + e];
    cs[e] = s;
  }
  __syncthreads();
  int o = threadIdx.x;   // PH = 256
  const float* w = Wout + ((size_t)(h * PH_) + o) * ENC_;
  float s = bout[h * PH_ + o];
  for (int e = 0; e < ENC_; ++e) s += cs[e] * w[e];
  context[(size_t)b * OUT_ + h * PH_ + o] = s;
}

// ---------------------------------------------------------------------------
extern "C" void kernel_launch(void* const* d_in, const int* in_sizes, int n_in,
                              void* d_out, int out_size, void* d_ws,
                              size_t ws_size, hipStream_t stream) {
  const float* enc        = (const float*)d_in[0];
  const int*   enc_len    = (const int*)d_in[1];
  const float* dec_states = (const float*)d_in[2];
  const float* Wenc       = (const float*)d_in[3];
  const float* benc       = (const float*)d_in[4];
  const float* Wdec       = (const float*)d_in[5];
  const float* bdec       = (const float*)d_in[6];
  const float* wattn      = (const float*)d_in[7];
  const float* Wconv      = (const float*)d_in[8];
  const float* Wloc       = (const float*)d_in[9];
  const float* bloc       = (const float*)d_in[10];
  const float* Wout       = (const float*)d_in[11];
  const float* bout       = (const float*)d_in[12];
  float* out = (float*)d_out;
  float* ws  = (float*)d_ws;

  // workspace layout (float offsets)
  float*  scoreP = ws;                         // 2*H*B*T   = 512000
  float*  attn   = ws + 512000;                // H*B*T     = 256000
  float*  Mtab   = ws + 768000;                // H*ATTN*102 = 208896
  float*  cumW   = ws + 976896;                // H*C*102   = 26112
  float*  decH   = ws + 1003008;               // H*B*ATTN  = 65536
  float*  ctxP   = ws + 1068544;               // TS*H*B*ENC = 1048576
  bf16_t* WencBF = (bf16_t*)(ws + 2117120);    // H*ATTN*ENC bf16 (4 MB)

  wencbf_kernel<<<(H_ * ATTN_ * ENC_) / (256 * 8), 256, 0, stream>>>(Wenc,
                                                                     WencBF);
  cumw_kernel<<<1, 256, 0, stream>>>(Wconv, cumW);
  mtab_kernel<<<H_ * ATTN_, 128, 0, stream>>>(cumW, Wloc, Mtab);
  dech_kernel<<<H_ * B_, 256, 0, stream>>>(dec_states, Wdec, bdec, decH);

  dim3 g4((T_ + 63) / 64, B_, H_ * 2);
  scores_kernel<<<g4, 256, 0, stream>>>(enc, enc_len, WencBF, benc, bloc,
                                        wattn, decH, Mtab, scoreP);

  softmax_kernel<<<H_ * B_, 256, 0, stream>>>(scoreP, enc_len, attn);
  attnmean_kernel<<<(B_ * T_ + 255) / 256, 256, 0, stream>>>(
      attn, out + (size_t)B_ * OUT_);

  dim3 g7(ENC_ / 256, B_, TS_);
  ctx_kernel<<<g7, 256, 0, stream>>>(enc, attn, ctxP);

  dim3 g8(H_, B_);
  out_kernel<<<g8, 256, 0, stream>>>(ctxP, Wout, bout, out);
}
